// TransformerBlock_23347442221593
// MI455X (gfx1250) — matrix operand order repaired and checked by reading, not compiled, not run on hardware
//
#include <hip/hip_runtime.h>
#include <hip/hip_bf16.h>
#include <math.h>
#include <stdint.h>

#define B_   2
#define S_   2048
#define D_   1024
#define H_   16
#define DH_  64
#define DFF_ 4096
#define MTOT (B_ * S_)      // 4096 rows
#define EPSF 1e-6f

typedef _Float16 h8   __attribute__((ext_vector_type(8)));
typedef _Float16 v16h __attribute__((ext_vector_type(16)));
typedef float    v8f  __attribute__((ext_vector_type(8)));

// Build a 16-half WMMA fragment from two 16-byte LDS/global chunks.
__device__ __forceinline__ v16h frag16(const _Float16* p0, const _Float16* p1) {
  h8 lo = *(const h8*)p0;
  h8 hi = *(const h8*)p1;
  return __builtin_shufflevector(lo, hi, 0,1,2,3,4,5,6,7,8,9,10,11,12,13,14,15);
}

// CDNA5 async global->LDS copy, 16B per lane, tracked by ASYNCcnt.
__device__ __forceinline__ void async_g2l_b128(const void* gptr, void* ldsptr) {
  uint32_t l = (uint32_t)(uintptr_t)ldsptr;
  uint64_t g = (uint64_t)(uintptr_t)gptr;
  asm volatile("global_load_async_to_lds_b128 %0, %1, off" :: "v"(l), "v"(g) : "memory");
}
__device__ __forceinline__ void wait_async0() {
  asm volatile("s_wait_asynccnt 0" ::: "memory");
}

// ds_swizzle XOR lane swap within each 16-lane half: offset = xor<<10 | and(0x1F)
#define SWZ(x, imm) __int_as_float(__builtin_amdgcn_ds_swizzle(__float_as_int(x), (imm)))
__device__ __forceinline__ float xor_max16(float x) {
  x = fmaxf(x, SWZ(x, 0x041F));
  x = fmaxf(x, SWZ(x, 0x081F));
  x = fmaxf(x, SWZ(x, 0x101F));
  x = fmaxf(x, SWZ(x, 0x201F));
  return x;
}
__device__ __forceinline__ float xor_sum16(float x) {
  x += SWZ(x, 0x041F);
  x += SWZ(x, 0x081F);
  x += SWZ(x, 0x101F);
  x += SWZ(x, 0x201F);
  return x;
}

__device__ __forceinline__ float gelu_tanh_f(float x) {
  const float c = 0.7978845608028654f;  // sqrt(2/pi)
  return 0.5f * x * (1.0f + tanhf(c * (x + 0.044715f * x * x * x)));
}

// ---------------------------------------------------------------------------
// Weight convert: f32 [K][N] row-major  ->  f16 [N][K] row-major (transposed)
// ---------------------------------------------------------------------------
__global__ void wcvt_kernel(const float* __restrict__ in, _Float16* __restrict__ out,
                            int K, int N) {
  long idx = (long)blockIdx.x * 256 + threadIdx.x;
  long tot = (long)K * N;
  if (idx >= tot) return;
  int n = (int)(idx / K);
  int k = (int)(idx % K);
  out[idx] = (_Float16)in[(long)k * N + n];
}

// ---------------------------------------------------------------------------
// LayerNorm (ddof=1 variance, eps added to std), f32 in -> f16 out
// ---------------------------------------------------------------------------
__global__ void ln_kernel(const float* __restrict__ x, const float* __restrict__ g,
                          const float* __restrict__ bb, _Float16* __restrict__ out) {
  int row = blockIdx.x;
  int tid = threadIdx.x;
  const float* xr = x + (long)row * D_;
  float4 v = *(const float4*)(xr + tid * 4);
  __shared__ float red[256];
  red[tid] = v.x + v.y + v.z + v.w;
  __syncthreads();
  for (int o = 128; o > 0; o >>= 1) { if (tid < o) red[tid] += red[tid + o]; __syncthreads(); }
  float mean = red[0] * (1.0f / D_);
  __syncthreads();
  float dx = v.x - mean, dy = v.y - mean, dz = v.z - mean, dw = v.w - mean;
  red[tid] = dx * dx + dy * dy + dz * dz + dw * dw;
  __syncthreads();
  for (int o = 128; o > 0; o >>= 1) { if (tid < o) red[tid] += red[tid + o]; __syncthreads(); }
  float var = red[0] * (1.0f / (D_ - 1));
  float rs = 1.0f / (sqrtf(var) + EPSF);
  float d4[4] = {dx, dy, dz, dw};
  for (int j = 0; j < 4; ++j) {
    int col = tid * 4 + j;
    out[(long)row * D_ + col] = (_Float16)(d4[j] * rs * g[col] + bb[col]);
  }
}

// ---------------------------------------------------------------------------
// Tiled WMMA GEMM with async double-buffered LDS pipeline.
// C[M,N] = A[M,K](f16) * Bt[N,K]^T(f16) + bias (+ resid)
// MODE 0: f16 out = acc + bias ; MODE 1: f32 out = acc+bias+resid ; MODE 2: gelu
// ---------------------------------------------------------------------------
#define BM  128
#define BN  128
#define BK  64
#define LDK 72

template <int MODE>
__global__ void gemm_kernel(const _Float16* __restrict__ A,
                            const _Float16* __restrict__ Bt,
                            const float* __restrict__ bias,
                            const float* __restrict__ resid,
                            _Float16* __restrict__ outh,
                            float* __restrict__ outf,
                            int M, int N, int K) {
  __shared__ _Float16 sA[2][BM * LDK];
  __shared__ _Float16 sB[2][BN * LDK];
  int tid = threadIdx.x;
  int lane = tid & 31, wid = tid >> 5;
  int lmod = lane & 15, lhalf = lane >> 4;
  int wm0 = (wid >> 1) * 32;   // 0,32,64,96
  int wn0 = (wid & 1) * 64;    // 0,64
  int m0 = blockIdx.y * BM;
  int n0 = blockIdx.x * BN;

  auto load_tile = [&](int buf, int k0) {
    for (int l = 0; l < 4; ++l) {
      int ch = tid + l * 256;        // 0..1023
      int r = ch >> 3;               // 0..127
      int c = (ch & 7) * 8;          // 0..56 halfs
      async_g2l_b128(&A[(long)(m0 + r) * K + k0 + c], &sA[buf][r * LDK + c]);
      async_g2l_b128(&Bt[(long)(n0 + r) * K + k0 + c], &sB[buf][r * LDK + c]);
    }
  };

  v8f acc[2][4] = {};

  load_tile(0, 0);
  wait_async0();
  __syncthreads();

  int buf = 0;
  for (int k0 = 0; k0 < K; k0 += BK) {
    if (k0 + BK < K) load_tile(buf ^ 1, k0 + BK);   // stream next tile async

    for (int kk = 0; kk < BK; kk += 32) {
      v16h af[2], bf[4];
      for (int mi = 0; mi < 2; ++mi) {
        const _Float16* p = &sA[buf][(wm0 + mi * 16 + lmod) * LDK + kk + lhalf * 8];
        af[mi] = frag16(p, p + 16);
      }
      for (int ni = 0; ni < 4; ++ni) {
        const _Float16* p = &sB[buf][(wn0 + ni * 16 + lmod) * LDK + kk + lhalf * 8];
        bf[ni] = frag16(p, p + 16);
      }
      for (int mi = 0; mi < 2; ++mi)
        for (int ni = 0; ni < 4; ++ni)
          acc[mi][ni] = __builtin_amdgcn_wmma_f32_16x16x32_f16(
              false, af[mi], false, bf[ni], (short)0, acc[mi][ni], false, false);
    }
    wait_async0();
    __syncthreads();
    buf ^= 1;
  }

  for (int mi = 0; mi < 2; ++mi)
    for (int ni = 0; ni < 4; ++ni) {
      int col = n0 + wn0 + ni * 16 + lmod;
      float bv = bias[col];
      for (int i = 0; i < 8; ++i) {
        int row = m0 + wm0 + mi * 16 + i + lhalf * 8;
        float v = acc[mi][ni][i] + bv;
        if (MODE == 0) {
          outh[(long)row * N + col] = (_Float16)v;
        } else if (MODE == 1) {
          outf[(long)row * N + col] = v + resid[(long)row * N + col];
        } else {
          outh[(long)row * N + col] = (_Float16)gelu_tanh_f(v);
        }
      }
    }
}

// ---------------------------------------------------------------------------
// Flash attention, 32-key chunks; Q tile staged in LDS (no persistent Q regs).
// block = (b, h, 128 q rows); 8 waves x 16 q rows each.
// ---------------------------------------------------------------------------
#define KC   32          // keys per chunk
#define LDP  40          // lds stride (halfs) for 32-wide key dimension

__global__ void attn_kernel(const _Float16* __restrict__ Q,
                            const _Float16* __restrict__ Km,
                            const _Float16* __restrict__ V,
                            const int* __restrict__ mask,
                            _Float16* __restrict__ out) {
  __shared__ _Float16 sQ[8][16 * LDK];       // per-wave Q tile [q][dh] (stride 72)
  __shared__ _Float16 sK[KC * LDK];          // [key][dh]   32 x 64 (stride 72)
  __shared__ _Float16 sVt[DH_ * LDP];        // [dh][key]   64 x 32 (stride 40)
  __shared__ _Float16 sP[8][16 * LDP];       // per-wave 16 x 32 probs
  __shared__ int smask[KC];

  int tid = threadIdx.x, lane = tid & 31, wid = tid >> 5;
  int lmod = lane & 15, lhalf = lane >> 4;
  int b = blockIdx.y >> 4;     // / H
  int h = blockIdx.y & 15;     // % H
  int q0 = blockIdx.x * 128 + wid * 16;

  // async-stage this wave's Q tile into LDS (16 rows x 64 halfs)
  for (int l = 0; l < 4; ++l) {
    int ch = lane + l * 32;      // 0..127
    int r = ch >> 3;             // q row 0..15
    int c = (ch & 7) * 8;        // dh offset
    long g = (long)(b * S_ + q0 + r) * D_ + h * DH_ + c;
    async_g2l_b128(&Q[g], &sQ[wid][r * LDK + c]);
  }

  float mrow[8], lrow[8];
  v8f o[4] = {};
  for (int i = 0; i < 8; ++i) { mrow[i] = -INFINITY; lrow[i] = 0.0f; }
  const float scale = 0.125f;  // 1/sqrt(64)

  for (int kb = 0; kb < S_; kb += KC) {
    // K chunk: async copy straight into LDS (row-major [key][dh])
    {
      int r = tid >> 3;            // key 0..31
      int c = (tid & 7) * 8;       // dh offset
      long g = (long)(b * S_ + kb + r) * D_ + h * DH_ + c;
      async_g2l_b128(&Km[g], &sK[r * LDK + c]);
      // V chunk transposed into sVt (through VGPRs)
      h8 vv = *(const h8*)&V[g];
      for (int j = 0; j < 8; ++j) sVt[(c + j) * LDP + r] = vv[j];
    }
    if (tid < KC) smask[tid] = mask[b * S_ + kb + tid];
    wait_async0();
    __syncthreads();

    // Q A-fragments from LDS (short live range)
    v16h qa[2];
    for (int c = 0; c < 2; ++c) {
      const _Float16* pq = &sQ[wid][lmod * LDK + c * 32 + lhalf * 8];
      qa[c] = frag16(pq, pq + 16);
    }

    // scores: Q(16x64) x K^T -> 2 tiles of 16 keys
    v8f st[2];
    for (int t = 0; t < 2; ++t) {
      v8f a = {};
      for (int c = 0; c < 2; ++c) {
        const _Float16* p = &sK[(t * 16 + lmod) * LDK + c * 32 + lhalf * 8];
        v16h bf = frag16(p, p + 16);
        a = __builtin_amdgcn_wmma_f32_16x16x32_f16(false, qa[c], false, bf,
                                                   (short)0, a, false, false);
      }
      bool ok = smask[t * 16 + lmod] != 0;
      for (int i = 0; i < 8; ++i) st[t][i] = ok ? a[i] * scale : -1e9f;
    }

    // online softmax (ds_swizzle XOR reductions over the 16-lane row group)
    float corr[8];
    for (int i = 0; i < 8; ++i) {
      float tmax = xor_max16(fmaxf(st[0][i], st[1][i]));
      float mnew = fmaxf(mrow[i], tmax);
      corr[i] = expf(mrow[i] - mnew);
      mrow[i] = mnew;
    }
    for (int i = 0; i < 8; ++i) {
      float p0 = expf(st[0][i] - mrow[i]);
      float p1 = expf(st[1][i] - mrow[i]);
      st[0][i] = p0; st[1][i] = p1;
      float s = xor_sum16(p0 + p1);
      lrow[i] = lrow[i] * corr[i] + s;
      for (int d = 0; d < 4; ++d) o[d][i] *= corr[i];
    }

    // stage P through LDS to re-layout C-frag -> A-frag
    _Float16* pw = sP[wid];
    for (int t = 0; t < 2; ++t)
      for (int i = 0; i < 8; ++i)
        pw[(i + lhalf * 8) * LDP + t * 16 + lmod] = (_Float16)st[t][i];
    __syncthreads();

    // O += P(16x32) x V(32x64)
    {
      const _Float16* pa = &pw[lmod * LDP + lhalf * 8];
      v16h af = frag16(pa, pa + 16);
      for (int d = 0; d < 4; ++d) {
        const _Float16* pb = &sVt[(d * 16 + lmod) * LDP + lhalf * 8];
        v16h bf = frag16(pb, pb + 16);
        o[d] = __builtin_amdgcn_wmma_f32_16x16x32_f16(false, af, false, bf,
                                                      (short)0, o[d], false, false);
      }
    }
    __syncthreads();
  }

  for (int d = 0; d < 4; ++d)
    for (int i = 0; i < 8; ++i) {
      int row = q0 + i + lhalf * 8;
      int col = h * DH_ + d * 16 + lmod;
      out[(long)(b * S_ + row) * D_ + col] = (_Float16)(o[d][i] / lrow[i]);
    }
}

// ---------------------------------------------------------------------------
extern "C" void kernel_launch(void* const* d_in, const int* in_sizes, int n_in,
                              void* d_out, int out_size, void* d_ws, size_t ws_size,
                              hipStream_t stream) {
  (void)in_sizes; (void)n_in; (void)out_size; (void)ws_size;
  const float* hidden = (const float*)d_in[0];
  const int*   mask   = (const int*)d_in[1];
  const float* wq = (const float*)d_in[2];
  const float* bq = (const float*)d_in[3];
  const float* wk = (const float*)d_in[4];
  const float* bk = (const float*)d_in[5];
  const float* wv = (const float*)d_in[6];
  const float* bv = (const float*)d_in[7];
  const float* wo = (const float*)d_in[8];
  const float* bo = (const float*)d_in[9];
  const float* w1 = (const float*)d_in[10];
  const float* b1 = (const float*)d_in[11];
  const float* w2 = (const float*)d_in[12];
  const float* b2 = (const float*)d_in[13];
  const float* ln1g = (const float*)d_in[14];
  const float* ln1b = (const float*)d_in[15];
  const float* ln2g = (const float*)d_in[16];
  const float* ln2b = (const float*)d_in[17];

  char* ws = (char*)d_ws;
  size_t off = 0;
  auto alloc = [&](size_t bytes) -> void* {
    void* p = ws + off;
    off += (bytes + 255) & ~(size_t)255;
    return p;
  };

  _Float16* wqt = (_Float16*)alloc((size_t)D_ * D_ * 2);
  _Float16* wkt = (_Float16*)alloc((size_t)D_ * D_ * 2);
  _Float16* wvt = (_Float16*)alloc((size_t)D_ * D_ * 2);
  _Float16* wot = (_Float16*)alloc((size_t)D_ * D_ * 2);
  _Float16* w1t = (_Float16*)alloc((size_t)D_ * DFF_ * 2);
  _Float16* w2t = (_Float16*)alloc((size_t)D_ * DFF_ * 2);
  _Float16* xln = (_Float16*)alloc((size_t)MTOT * D_ * 2);
  _Float16* qh  = (_Float16*)alloc((size_t)MTOT * D_ * 2);
  _Float16* kh  = (_Float16*)alloc((size_t)MTOT * D_ * 2);
  _Float16* vh  = (_Float16*)alloc((size_t)MTOT * D_ * 2);
  _Float16* hid = (_Float16*)alloc((size_t)MTOT * D_ * 2);
  float*    hres= (float*)   alloc((size_t)MTOT * D_ * 4);
  _Float16* ffin= (_Float16*)alloc((size_t)MTOT * D_ * 2);
  _Float16* act = (_Float16*)alloc((size_t)MTOT * DFF_ * 2);

  // weights -> f16, transposed to [N][K]
  int gsq = (D_ * D_ + 255) / 256;
  wcvt_kernel<<<gsq, 256, 0, stream>>>(wq, wqt, D_, D_);
  wcvt_kernel<<<gsq, 256, 0, stream>>>(wk, wkt, D_, D_);
  wcvt_kernel<<<gsq, 256, 0, stream>>>(wv, wvt, D_, D_);
  wcvt_kernel<<<gsq, 256, 0, stream>>>(wo, wot, D_, D_);
  int gff = (D_ * DFF_ + 255) / 256;
  wcvt_kernel<<<gff, 256, 0, stream>>>(w1, w1t, D_, DFF_);   // K=D, N=DFF
  wcvt_kernel<<<gff, 256, 0, stream>>>(w2, w2t, DFF_, D_);   // K=DFF, N=D

  // LN1
  ln_kernel<<<MTOT, 256, 0, stream>>>(hidden, ln1g, ln1b, xln);

  // QKV projections
  dim3 g1(D_ / BN, MTOT / BM);           // (8, 32)
  gemm_kernel<0><<<g1, 256, 0, stream>>>(xln, wqt, bq, nullptr, qh, nullptr, MTOT, D_, D_);
  gemm_kernel<0><<<g1, 256, 0, stream>>>(xln, wkt, bk, nullptr, kh, nullptr, MTOT, D_, D_);
  gemm_kernel<0><<<g1, 256, 0, stream>>>(xln, wvt, bv, nullptr, vh, nullptr, MTOT, D_, D_);

  // attention
  dim3 ga(S_ / 128, B_ * H_);            // (16, 32)
  attn_kernel<<<ga, 256, 0, stream>>>(qh, kh, vh, mask, hid);

  // out-proj + residual -> hres (f32)
  gemm_kernel<1><<<g1, 256, 0, stream>>>(hid, wot, bo, hidden, nullptr, hres, MTOT, D_, D_);

  // LN2
  ln_kernel<<<MTOT, 256, 0, stream>>>(hres, ln2g, ln2b, ffin);

  // FFN1 with GELU -> act (f16)
  dim3 g2(DFF_ / BN, MTOT / BM);         // (32, 32)
  gemm_kernel<2><<<g2, 256, 0, stream>>>(ffin, w1t, b1, nullptr, act, nullptr, MTOT, DFF_, D_);

  // FFN2 + residual -> d_out (f32)
  gemm_kernel<1><<<g1, 256, 0, stream>>>(act, w2t, b2, hres, nullptr, (float*)d_out, MTOT, D_, DFF_);
}
